// DynamicConv_24970939859163
// MI455X (gfx1250) — matrix operand order repaired and checked by reading, not compiled, not run on hardware
//
#include <hip/hip_runtime.h>
#include <hip/hip_bf16.h>

typedef __bf16 bf16_t;
typedef __bf16 v16bf __attribute__((ext_vector_type(16)));
typedef __bf16 v8bf  __attribute__((ext_vector_type(8)));
typedef float  v8f   __attribute__((ext_vector_type(8)));
typedef unsigned int u32x4 __attribute__((ext_vector_type(4)));
typedef int i32x8 __attribute__((ext_vector_type(8)));
typedef int i32x4 __attribute__((ext_vector_type(4)));

constexpr int kN     = 4096;
constexpr int kCin   = 256;
constexpr int kCfeat = 64;
constexpr int kCout  = 256;
constexpr int kHW    = 49;
constexpr int kSPad  = 64;               // 49 spatial rows padded to 64
constexpr int kNpIn  = kCin * kCfeat;    // 16384
constexpr int kNpTot = 2 * kNpIn;        // 32768
constexpr int kKfc   = kHW * kCout;      // 12544
constexpr float kEps = 1e-5f;

// Tensor Data Mover available? (device pass only; this toolchain = 6-arg form)
#if defined(__gfx1250__) && __has_builtin(__builtin_amdgcn_tensor_load_to_lds) && \
    __has_builtin(__builtin_amdgcn_s_wait_tensorcnt)
#define USE_TDM 1
#else
#define USE_TDM 0
#endif

// ---------------------------------------------------------------------------
// A-fragment (16x32 bf16, row-major source): lane L holds row m=L&15.
// VGPR0-7 <- K = h*16 .. h*16+15  (h = L>>4)
// ---------------------------------------------------------------------------
__device__ inline v16bf load_a_frag(const bf16_t* row, int hoff) {
  v16bf a = *(const v16bf*)(row + hoff * 2);
  return a;
}

#if USE_TDM
// ---------------------------------------------------------------------------
// TDM: DMA a contiguous block of `nelems` bf16 from global to LDS.
// D# group0: count=1, lds_addr, 57-bit global addr, type=2 (bits 127:126).
// D# group1: data_size=1 (2B); 1-D tile: tensor_dim0=tile_dim0=stride=nelems,
//            tensor_dim1=tile_dim1=1. Groups 2/3 zero (<=2D tensor).
// Issued once by one wave; tracked by TENSORcnt.
// ---------------------------------------------------------------------------
__device__ inline void tdm_copy_to_lds(void* lds, const void* gsrc, int nelems) {
  uint32_t ldsA = (uint32_t)(uintptr_t)lds;        // low 32 bits = LDS offset
  uint64_t ga = (uint64_t)(uintptr_t)gsrc;
  u32x4 g0 = { 1u, ldsA, (uint32_t)ga,
               (uint32_t)((ga >> 32) & 0x01FFFFFFu) | 0x80000000u };
  i32x8 g1 = { (int)(1u << 16),                        // data_size=1 (2B)
               (int)((nelems & 0xFFFF) << 16),         // tensor_dim0 lo16
               (int)(((nelems >> 16) & 0xFFFF) | (1 << 16)), // td0 hi | td1 lo=1
               (int)((nelems & 0xFFFF) << 16),         // td1 hi=0 | tile_dim0
               1,                                      // tile_dim1=1, tile_dim2=0
               nelems,                                 // tensor_dim0_stride lo32
               0, 0 };
  i32x4 z4 = { 0, 0, 0, 0 };
  i32x8 z8 = { 0, 0, 0, 0, 0, 0, 0, 0 };
  __builtin_amdgcn_tensor_load_to_lds(g0, g1, z4, z4, z8, 0);
}
#endif

// ---------------------------------------------------------------------------
// K0: fp32 -> bf16 conversion (grid-stride)
// ---------------------------------------------------------------------------
__global__ void cvt_f32_bf16(const float* __restrict__ src,
                             bf16_t* __restrict__ dst, int n) {
  int stride = gridDim.x * blockDim.x;
  for (int i = blockIdx.x * blockDim.x + threadIdx.x; i < n; i += stride)
    dst[i] = (bf16_t)src[i];
}

// ---------------------------------------------------------------------------
// K0b: fp32 (R x C row-major) -> bf16 transposed (C x R). 64x64 LDS tiles,
// coalesced on both sides. Used for W_dyn^T and W_fc^T (one-time cost).
// ---------------------------------------------------------------------------
__global__ void xpose_cvt(const float* __restrict__ src,
                          bf16_t* __restrict__ dst, int R, int C) {
  __shared__ bf16_t tile[64][65];
  const int c0 = blockIdx.x * 64, r0 = blockIdx.y * 64;
  const int tc = threadIdx.x & 63;
  const int tr4 = threadIdx.x >> 6;  // 0..3
#pragma unroll
  for (int i = 0; i < 16; ++i) {
    int r = tr4 + i * 4;
    tile[r][tc] = (bf16_t)src[(size_t)(r0 + r) * C + (c0 + tc)];
  }
  __syncthreads();
#pragma unroll
  for (int i = 0; i < 16; ++i) {
    int c = tr4 + i * 4;
    dst[(size_t)(c0 + c) * R + (r0 + tc)] = tile[tc][c];
  }
}

// ---------------------------------------------------------------------------
// K1: input_feature (N, 256, 49) fp32 -> pX (N, 64, 256) bf16 (transposed,
// spatial padded to 64 rows with zeros). LDS-staged so global reads coalesce.
// ---------------------------------------------------------------------------
__global__ void xpose_kernel(const float* __restrict__ x,
                             bf16_t* __restrict__ pX) {
  __shared__ bf16_t tile[kCin * kHW];  // 25 KB
  const int n = blockIdx.x;
  const float* src = x + (size_t)n * kCin * kHW;
  for (int i = threadIdx.x; i < kCin * kHW; i += blockDim.x)
    tile[i] = (bf16_t)src[i];
  __syncthreads();
  bf16_t* dst = pX + (size_t)n * kSPad * kCin;
  for (int i = threadIdx.x; i < kSPad * kCin; i += blockDim.x) {
    int s = i >> 8, c = i & 255;
    dst[i] = (s < kHW) ? tile[c * kHW + s] : (bf16_t)0.0f;
  }
}

// ---------------------------------------------------------------------------
// K2: params = pA(4096x256) @ W_dyn + b_dyn, with W_dyn pre-transposed so each
// lane's B fragment is one contiguous 32B global read (L2-resident, 16.8MB).
// Epilogue scatters into pP with p_in/p_out PRE-TRANSPOSED per proposal:
//   p_inT  [f(64)][c(256)]  at pP[n][f*256+c]
//   p_outT [o(256)][f(64)]  at pP[n][16384 + o*64+f]
// so K3/K4 B staging is a contiguous block copy. No LDS, no barriers here.
// ---------------------------------------------------------------------------
__global__ void params_gemm(const bf16_t* __restrict__ pA,
                            const bf16_t* __restrict__ pWT,
                            const float* __restrict__ b_dyn,
                            bf16_t* __restrict__ pP) {
  const int m0 = blockIdx.x * 16;
  const int col0 = blockIdx.y * 128;
  const int t = threadIdx.x;
  const int lane = t & 31, wave = t >> 5;
  const int lm = lane & 15, lh = lane >> 4;
  const int col = col0 + wave * 16 + lm;
  const bf16_t* arow = pA + (size_t)(m0 + lm) * kCin;
  const bf16_t* bcol = pWT + (size_t)col * kCin;  // [col][k]
  v8f acc = {};
#pragma unroll
  for (int k0 = 0; k0 < kCin; k0 += 32) {
    v16bf a = load_a_frag(arow + k0, lh * 8);
    v16bf b = *(const v16bf*)(bcol + k0 + lh * 16);
    acc = __builtin_amdgcn_wmma_f32_16x16x32_bf16(false, a, false, b,
                                                  (short)0, acc, false, false);
  }
  const float bias = b_dyn[col];
  int pidx;
  if (col < kNpIn) {                       // p_in: j = c*64+f -> [f][c]
    pidx = (col & 63) * kCin + (col >> 6);
  } else {                                 // p_out: jj = f*256+o -> [o][f]
    int jj = col - kNpIn;
    pidx = kNpIn + (jj & 255) * kCfeat + (jj >> 8);
  }
#pragma unroll
  for (int v = 0; v < 8; ++v) {
    int row = m0 + v + lh * 8;
    pP[(size_t)row * kNpTot + pidx] = (bf16_t)(acc[v] + bias);
  }
}

// ---------------------------------------------------------------------------
// K3: per proposal, F1 = relu(LN(X(64x256) @ p_in(256x64))) -> pF1 bf16.
// p_inT is already [col][k] in global -> TDM block-copies 32KB into LDS.
// Row LN stats via LDS float atomics straight from the WMMA accumulators.
// ---------------------------------------------------------------------------
__global__ void bmm1_ln_relu(const bf16_t* __restrict__ pX,
                             const bf16_t* __restrict__ pP,
                             const float* __restrict__ g_in,
                             const float* __restrict__ be_in,
                             bf16_t* __restrict__ pF1) {
  __shared__ __align__(32) bf16_t ldsB[kCfeat * kCin];  // [col(64)][k(256)] 32KB
  __shared__ float rsum[kSPad], rsq[kSPad], rmu[kSPad], rrs[kSPad];
  const int n = blockIdx.x;
  const int t = threadIdx.x;
  const int lane = t & 31, wave = t >> 5;
  const int lm = lane & 15, lh = lane >> 4;
  const bf16_t* bsrc = pP + (size_t)n * kNpTot;  // p_inT, contiguous 32KB
#if USE_TDM
  if (wave == 0) {
    tdm_copy_to_lds(ldsB, bsrc, kNpIn);
    __builtin_amdgcn_s_wait_tensorcnt(0);
  }
#else
  for (int i = t * 8; i < kNpIn; i += 256 * 8)
    *(v8bf*)(ldsB + i) = *(const v8bf*)(bsrc + i);
#endif
  if (t < kSPad) { rsum[t] = 0.f; rsq[t] = 0.f; }
  __syncthreads();

  const bf16_t* A = pX + (size_t)n * kSPad * kCin;
  const int mt = wave >> 1;          // tiles 2w, 2w+1 share the M tile
  const int nt0 = (wave & 1) * 2;
  v8f acc0 = {}, acc1 = {};
#pragma unroll
  for (int k0 = 0; k0 < kCin; k0 += 32) {
    v16bf a = load_a_frag(A + (size_t)(mt * 16 + lm) * kCin + k0, lh * 8);
    v16bf b0 = *(const v16bf*)(&ldsB[(nt0 * 16 + lm) * kCin + k0 + lh * 16]);
    v16bf b1 = *(const v16bf*)(&ldsB[((nt0 + 1) * 16 + lm) * kCin + k0 + lh * 16]);
    acc0 = __builtin_amdgcn_wmma_f32_16x16x32_bf16(false, a, false, b0,
                                                   (short)0, acc0, false, false);
    acc1 = __builtin_amdgcn_wmma_f32_16x16x32_bf16(false, a, false, b1,
                                                   (short)0, acc1, false, false);
  }
#pragma unroll
  for (int v = 0; v < 8; ++v) {
    int row = mt * 16 + v + lh * 8;
    atomicAdd(&rsum[row], acc0[v] + acc1[v]);
    atomicAdd(&rsq[row], acc0[v] * acc0[v] + acc1[v] * acc1[v]);
  }
  __syncthreads();
  if (t < kSPad) {
    float mu = rsum[t] * (1.0f / kCfeat);
    float var = rsq[t] * (1.0f / kCfeat) - mu * mu;
    rmu[t] = mu;
    rrs[t] = rsqrtf(var + kEps);
  }
  __syncthreads();
  const int col0 = nt0 * 16 + lm;
  const float g0 = g_in[col0], b0v = be_in[col0];
  const float g1 = g_in[col0 + 16], b1v = be_in[col0 + 16];
  bf16_t* out = pF1 + (size_t)n * kSPad * kCfeat;
#pragma unroll
  for (int v = 0; v < 8; ++v) {
    int row = mt * 16 + v + lh * 8;
    float x0 = fmaxf((acc0[v] - rmu[row]) * rrs[row] * g0 + b0v, 0.f);
    float x1 = fmaxf((acc1[v] - rmu[row]) * rrs[row] * g1 + b1v, 0.f);
    out[row * kCfeat + col0] = (bf16_t)x0;
    out[row * kCfeat + col0 + 16] = (bf16_t)x1;
  }
}

// ---------------------------------------------------------------------------
// K4: per proposal, F2 = relu(LN(F1(64x64) @ p_out(64x256))) -> pF2 bf16
// (stored unpadded (N,49*256) so the fc K dim is contiguous).
// p_outT already [col][k] in global -> TDM block-copy 32KB into LDS.
// ---------------------------------------------------------------------------
__global__ void bmm2_ln_relu(const bf16_t* __restrict__ pF1,
                             const bf16_t* __restrict__ pP,
                             const float* __restrict__ g_out,
                             const float* __restrict__ be_out,
                             bf16_t* __restrict__ pF2) {
  __shared__ __align__(32) bf16_t ldsB[kCout * kCfeat];  // [col(256)][k(64)] 32KB
  __shared__ float rsum[kSPad], rsq[kSPad], rmu[kSPad], rrs[kSPad];
  const int n = blockIdx.x;
  const int t = threadIdx.x;
  const int lane = t & 31, wave = t >> 5;
  const int lm = lane & 15, lh = lane >> 4;
  const bf16_t* bsrc = pP + (size_t)n * kNpTot + kNpIn;  // p_outT, 32KB
#if USE_TDM
  if (wave == 0) {
    tdm_copy_to_lds(ldsB, bsrc, kNpIn);
    __builtin_amdgcn_s_wait_tensorcnt(0);
  }
#else
  for (int i = t * 8; i < kNpIn; i += 256 * 8)
    *(v8bf*)(ldsB + i) = *(const v8bf*)(bsrc + i);
#endif
  if (t < kSPad) { rsum[t] = 0.f; rsq[t] = 0.f; }
  __syncthreads();

  const bf16_t* A = pF1 + (size_t)n * kSPad * kCfeat;
  const int mt = wave >> 1;
  const int ntb = (wave & 1) * 8;  // 8 N-tiles per wave
  v8f acc[8] = {};
#pragma unroll
  for (int k0 = 0; k0 < kCfeat; k0 += 32) {
    v16bf a = load_a_frag(A + (size_t)(mt * 16 + lm) * kCfeat + k0, lh * 8);
#pragma unroll
    for (int j = 0; j < 8; ++j) {
      int col = (ntb + j) * 16 + lm;
      v16bf b = *(const v16bf*)(&ldsB[col * kCfeat + k0 + lh * 16]);
      acc[j] = __builtin_amdgcn_wmma_f32_16x16x32_bf16(
          false, a, false, b, (short)0, acc[j], false, false);
    }
  }
#pragma unroll
  for (int v = 0; v < 8; ++v) {
    int row = mt * 16 + v + lh * 8;
    float s = 0.f, q = 0.f;
#pragma unroll
    for (int j = 0; j < 8; ++j) { s += acc[j][v]; q += acc[j][v] * acc[j][v]; }
    atomicAdd(&rsum[row], s);
    atomicAdd(&rsq[row], q);
  }
  __syncthreads();
  if (t < kSPad) {
    float mu = rsum[t] * (1.0f / kCout);
    float var = rsq[t] * (1.0f / kCout) - mu * mu;
    rmu[t] = mu;
    rrs[t] = rsqrtf(var + kEps);
  }
  __syncthreads();
  bf16_t* out = pF2 + (size_t)n * kKfc;
#pragma unroll
  for (int j = 0; j < 8; ++j) {
    int col = (ntb + j) * 16 + lm;
    float g = g_out[col], be = be_out[col];
#pragma unroll
    for (int v = 0; v < 8; ++v) {
      int row = mt * 16 + v + lh * 8;
      if (row < kHW) {
        float x = fmaxf((acc[j][v] - rmu[row]) * rrs[row] * g + be, 0.f);
        out[row * kCout + col] = (bf16_t)x;
      }
    }
  }
}

// ---------------------------------------------------------------------------
// K5: out = relu(LN(F2flat(4096x12544) @ W_fc + b_fc)) fp32.
// W_fc pre-transposed -> B fragments are contiguous 32B global reads from the
// L2-resident 6.4MB W_fc^T; no LDS B, no barriers in the 392-step K loop.
// 512 threads = 16 waves cover the full 256-col row, so LN stays WG-local.
// ---------------------------------------------------------------------------
__global__ void fc_ln_relu(const bf16_t* __restrict__ pF2,
                           const bf16_t* __restrict__ pWfcT,
                           const float* __restrict__ b_fc,
                           const float* __restrict__ g_fc,
                           const float* __restrict__ be_fc,
                           float* __restrict__ out) {
  __shared__ float rsum[16], rsq[16], rmu[16], rrs[16];
  const int m0 = blockIdx.x * 16;
  const int t = threadIdx.x;
  const int lane = t & 31, wave = t >> 5;  // 16 waves
  const int lm = lane & 15, lh = lane >> 4;
  const int col = wave * 16 + lm;
  const bf16_t* arow = pF2 + (size_t)(m0 + lm) * kKfc;
  const bf16_t* bcol = pWfcT + (size_t)col * kKfc;
  if (t < 16) { rsum[t] = 0.f; rsq[t] = 0.f; }

  v8f acc = {};
#pragma unroll 4
  for (int k0 = 0; k0 < kKfc; k0 += 32) {
    __builtin_prefetch(arow + k0 + 512, 0, 1);  // global_prefetch for A stream
    v16bf a = load_a_frag(arow + k0, lh * 8);
    v16bf b = *(const v16bf*)(bcol + k0 + lh * 16);
    acc = __builtin_amdgcn_wmma_f32_16x16x32_bf16(false, a, false, b,
                                                  (short)0, acc, false, false);
  }
  const float bias = b_fc[col];
#pragma unroll
  for (int v = 0; v < 8; ++v) acc[v] += bias;
  __syncthreads();
#pragma unroll
  for (int v = 0; v < 8; ++v) {
    int row = v + lh * 8;
    atomicAdd(&rsum[row], acc[v]);
    atomicAdd(&rsq[row], acc[v] * acc[v]);
  }
  __syncthreads();
  if (t < 16) {
    float mu = rsum[t] * (1.0f / kCout);
    float var = rsq[t] * (1.0f / kCout) - mu * mu;
    rmu[t] = mu;
    rrs[t] = rsqrtf(var + kEps);
  }
  __syncthreads();
  const float g = g_fc[col], be = be_fc[col];
#pragma unroll
  for (int v = 0; v < 8; ++v) {
    int row = v + lh * 8;
    float x = fmaxf((acc[v] - rmu[row]) * rrs[row] * g + be, 0.f);
    out[(size_t)(m0 + row) * kCout + col] = x;
  }
}

// ---------------------------------------------------------------------------
extern "C" void kernel_launch(void* const* d_in, const int* in_sizes, int n_in,
                              void* d_out, int out_size, void* d_ws,
                              size_t ws_size, hipStream_t stream) {
  const float* param_feature = (const float*)d_in[0];
  const float* input_feature = (const float*)d_in[1];
  const float* W_dyn = (const float*)d_in[2];
  const float* b_dyn = (const float*)d_in[3];
  const float* g_in  = (const float*)d_in[4];
  const float* be_in = (const float*)d_in[5];
  const float* g_out = (const float*)d_in[6];
  const float* be_out= (const float*)d_in[7];
  const float* W_fc  = (const float*)d_in[8];
  const float* b_fc  = (const float*)d_in[9];
  const float* g_fc  = (const float*)d_in[10];
  const float* be_fc = (const float*)d_in[11];
  float* out = (float*)d_out;

  char* ws = (char*)d_ws;
  size_t off = 0;
  auto carve = [&](size_t bytes) {
    size_t o = off;
    off = (off + bytes + 255) & ~(size_t)255;
    return o;
  };
  bf16_t* pA    = (bf16_t*)(ws + carve((size_t)kN * kCin * 2));          //   2 MB
  bf16_t* pWT   = (bf16_t*)(ws + carve((size_t)kCin * kNpTot * 2));      //  17 MB
  bf16_t* pWfcT = (bf16_t*)(ws + carve((size_t)kKfc * kCout * 2));       // 6.4 MB
  bf16_t* pX    = (bf16_t*)(ws + carve((size_t)kN * kSPad * kCin * 2));  // 134 MB
  bf16_t* pP    = (bf16_t*)(ws + carve((size_t)kN * kNpTot * 2));        // 268 MB
  bf16_t* pF1   = (bf16_t*)(ws + carve((size_t)kN * kSPad * kCfeat * 2));// 34 MB
  bf16_t* pF2   = (bf16_t*)(ws + carve((size_t)kN * kKfc * 2));          // 103 MB

  cvt_f32_bf16<<<1024, 256, 0, stream>>>(param_feature, pA, kN * kCin);
  // W_dyn (256 x 32768) -> pWT (32768 x 256)
  xpose_cvt<<<dim3(kNpTot / 64, kCin / 64), 256, 0, stream>>>(W_dyn, pWT,
                                                              kCin, kNpTot);
  // W_fc (12544 x 256) -> pWfcT (256 x 12544)
  xpose_cvt<<<dim3(kCout / 64, kKfc / 64), 256, 0, stream>>>(W_fc, pWfcT,
                                                             kKfc, kCout);
  xpose_kernel<<<kN, 256, 0, stream>>>(input_feature, pX);

  params_gemm<<<dim3(kN / 16, kNpTot / 128), 256, 0, stream>>>(pA, pWT, b_dyn,
                                                               pP);
  bmm1_ln_relu<<<kN, 256, 0, stream>>>(pX, pP, g_in, be_in, pF1);
  bmm2_ln_relu<<<kN, 256, 0, stream>>>(pF1, pP, g_out, be_out, pF2);
  fc_ln_relu<<<kN / 16, 512, 0, stream>>>(pF2, pWfcT, b_fc, g_fc, be_fc, out);
}
